// LRUBlock_7713761264351
// MI455X (gfx1250) — matrix operand order repaired and checked by reading, not compiled, not run on hardware
//
#include <hip/hip_runtime.h>
#include <math.h>

#define H_DIM 256
#define FFN_DIM 1024
#define B_DIM 32
#define T_DIM 2048
#define NROWS (B_DIM * T_DIM)           /* 65536 */
#define NX ((size_t)NROWS * H_DIM)      /* 16,777,216 */
#define LN_EPS 1e-5f
#define HPAD 264                         /* padded LDS row stride (halves), 256-col tiles */
#define ZPAD 1032                        /* padded LDS row stride (halves), 1024-col tile */

typedef __attribute__((ext_vector_type(16))) _Float16 v16h;
typedef __attribute__((ext_vector_type(8)))  _Float16 v8h;
typedef __attribute__((ext_vector_type(8)))  float    v8f;
typedef int v4i_vs __attribute__((vector_size(16)));   /* matches builtin param type */

union V16U { v16h v; v8h h[2]; };

// ---- async global -> LDS copy (CDNA5 GLOBAL_LOAD_ASYNC_TO_LDS_B128) -----
// Signature (from compiler diagnostic): (AS1 int4*, AS3 int4*, int offset, int cpol).
static __device__ __forceinline__ void cp_async_b128(_Float16* ldst,
                                                     const _Float16* gsrc) {
#if __has_builtin(__builtin_amdgcn_global_load_async_to_lds_b128)
  __builtin_amdgcn_global_load_async_to_lds_b128(
      (__attribute__((address_space(1))) v4i_vs*)(uintptr_t)gsrc,
      (__attribute__((address_space(3))) v4i_vs*)(unsigned)(uintptr_t)ldst,
      0, 0);
#else
  *(v8h*)ldst = *(const v8h*)gsrc;
#endif
}
static __device__ __forceinline__ void cp_async_wait() {
#if __has_builtin(__builtin_amdgcn_global_load_async_to_lds_b128)
#if __has_builtin(__builtin_amdgcn_s_wait_asynccnt)
  __builtin_amdgcn_s_wait_asynccnt(0);
#else
  asm volatile("s_wait_asynccnt 0" ::: "memory");
#endif
#endif
}

// ---- WMMA fragment loaders (wave32, 16x16x32 f16) ----------------------
// A (16x32, MxK): lane m=lane&15, hlf=lane>>4. elements 0..7 = K[k0+hlf*8 ..),
// elements 8..15 = K[k0+16+hlf*8 ..). Two contiguous 16B runs.
static __device__ __forceinline__ v16h ld_afrag(const _Float16* A, int stride,
                                                int m, int k0, int hlf) {
  V16U u;
  const _Float16* r = A + m * stride + k0 + hlf * 8;
  u.h[0] = *(const v8h*)(r);
  u.h[1] = *(const v8h*)(r + 16);
  return u.v;
}
// B (32x16, KxN) with B[k][n] = W[n][k]: lane n=lane&15, kh=lane>>4.
// elements 0..7 = K[k0+kh*8 ..), elements 8..15 = K[k0+16+kh*8 ..) along row n of W.
static __device__ __forceinline__ v16h ld_bfrag(const _Float16* W, int stride,
                                                int n, int k0, int kh) {
  V16U u;
  const _Float16* r = W + n * stride + k0 + kh * 8;
  u.h[0] = *(const v8h*)(r);
  u.h[1] = *(const v8h*)(r + 16);
  return u.v;
}

#define WMMA_F16(a, b, c) \
  __builtin_amdgcn_wmma_f32_16x16x32_f16(false, (a), false, (b), (short)0, (c), false, false)

// Split-f16 3-pass GEMM accumulate over K = KC*32, NT output tiles of 16 cols.
template <int NT, int KC>
static __device__ __forceinline__ void gemm_acc(
    v8f* acc,
    const _Float16* Ah, const _Float16* Al, int strideA,
    const _Float16* __restrict__ Wh, const _Float16* __restrict__ Wl, int strideB,
    int ncol_base, int lane) {
  const int m = lane & 15, hlf = lane >> 4;
  for (int kc = 0; kc < KC; ++kc) {
    const int k0 = kc * 32;
    v16h ah = ld_afrag(Ah, strideA, m, k0, hlf);
    v16h al = ld_afrag(Al, strideA, m, k0, hlf);
#pragma unroll
    for (int nt = 0; nt < NT; ++nt) {
      const int n = ncol_base + nt * 16 + m;
      v16h bh = ld_bfrag(Wh, strideB, n, k0, hlf);
      v16h bl = ld_bfrag(Wl, strideB, n, k0, hlf);
      acc[nt] = WMMA_F16(ah, bh, acc[nt]);
      acc[nt] = WMMA_F16(ah, bl, acc[nt]);
      acc[nt] = WMMA_F16(al, bh, acc[nt]);
    }
  }
}

// Stage a 16x256 f16 plane into LDS with padded row stride HPAD.
// 512 16B chunks, 256 threads -> 2 async copies per thread per plane.
static __device__ __forceinline__ void stage_plane(_Float16* s,
                                                   const _Float16* g, int tid) {
#pragma unroll
  for (int i = 0; i < 2; ++i) {
    const int c = tid + i * 256;            // chunk id 0..511
    const int r = c >> 5, c8 = (c & 31) * 8;
    cp_async_b128(s + r * HPAD + c8, g + r * 256 + c8);
  }
}

// ---- f32 -> f16 hi/lo split ---------------------------------------------
__global__ __launch_bounds__(256) void k_split(const float* __restrict__ src,
                                               _Float16* __restrict__ hi,
                                               _Float16* __restrict__ lo,
                                               int n, float scale) {
  int i = blockIdx.x * 256 + threadIdx.x;
  if (i >= n) return;
  float v = src[i] * scale;
  _Float16 h = (_Float16)v;
  hi[i] = h;
  lo[i] = (_Float16)(v - (float)h);
}

// ---- bx_re / bx_im = x @ B_re^T, x @ B_im^T -----------------------------
__global__ __launch_bounds__(256) void k_gemm_bx(
    const _Float16* __restrict__ xh, const _Float16* __restrict__ xl,
    const _Float16* __restrict__ Breh, const _Float16* __restrict__ Brel,
    const _Float16* __restrict__ Bimh, const _Float16* __restrict__ Biml,
    float* __restrict__ bx_re, float* __restrict__ bx_im) {
  __shared__ __align__(16) _Float16 sxh[16 * HPAD];
  __shared__ __align__(16) _Float16 sxl[16 * HPAD];
  const int tid = threadIdx.x;
  const int row0 = blockIdx.x * 16;
  stage_plane(sxh, xh + (size_t)row0 * 256, tid);
  stage_plane(sxl, xl + (size_t)row0 * 256, tid);
  // pull weight planes toward L2/WGP while the stage is in flight
  __builtin_prefetch(Breh + (size_t)tid * 128, 0, 1);
  __builtin_prefetch(Bimh + (size_t)tid * 128, 0, 1);
  cp_async_wait();
  __syncthreads();
  const int wave = tid >> 5, lane = tid & 31;
  const int hlf = lane >> 4, m = lane & 15;
  const bool im = wave >= 4;
  const _Float16* Wh = im ? Bimh : Breh;
  const _Float16* Wl = im ? Biml : Brel;
  float* out = im ? bx_im : bx_re;
  const int ncol0 = (wave & 3) * 64;
  v8f acc[4] = {};
  gemm_acc<4, 8>(acc, sxh, sxl, HPAD, Wh, Wl, 256, ncol0, lane);
#pragma unroll
  for (int nt = 0; nt < 4; ++nt) {
    const int c = ncol0 + nt * 16 + m;
#pragma unroll
    for (int v = 0; v < 8; ++v)
      out[(size_t)(row0 + v + 8 * hlf) * 256 + c] = acc[nt][v];
  }
}

// ---- diagonal complex scan: 8192 independent channels -------------------
__global__ __launch_bounds__(256) void k_scan(
    const float* __restrict__ bx_re, const float* __restrict__ bx_im,
    const float* __restrict__ nu_log, const float* __restrict__ theta_log,
    const float* __restrict__ gamma_log,
    _Float16* __restrict__ hrh, _Float16* __restrict__ hrl,
    _Float16* __restrict__ hih, _Float16* __restrict__ hil) {
  const int tid = blockIdx.x * 256 + threadIdx.x;  // 0..8191
  const int b = tid >> 8, ch = tid & 255;
  const float nu = expf(nu_log[ch]);
  const float th = expf(theta_log[ch]);
  const float er = expf(-nu);
  const float lr = er * cosf(th);
  const float li = er * sinf(th);
  const float g = expf(gamma_log[ch]);
  float hr = 0.f, hi = 0.f;
  size_t idx = (size_t)b * T_DIM * 256 + ch;
  float br = bx_re[idx], bi = bx_im[idx];        // 1-deep pipeline
  for (int t = 0; t < T_DIM - 1; ++t) {
    const size_t cur = idx;
    idx += 256;
    __builtin_prefetch(bx_re + cur + 16 * 256, 0, 1);
    __builtin_prefetch(bx_im + cur + 16 * 256, 0, 1);
    const float brn = bx_re[idx];
    const float bin = bx_im[idx];
    const float hrx = lr * hr - li * hi + g * br;  // faithful: h_im uses updated h_re
    const float hix = lr * hi + li * hrx + g * bi;
    hr = hrx; hi = hix;
    _Float16 a = (_Float16)hr;
    hrh[cur] = a; hrl[cur] = (_Float16)(hr - (float)a);
    _Float16 c = (_Float16)hi;
    hih[cur] = c; hil[cur] = (_Float16)(hi - (float)c);
    br = brn; bi = bin;
  }
  {  // epilogue (t = T-1)
    const float hrx = lr * hr - li * hi + g * br;
    const float hix = lr * hi + li * hrx + g * bi;
    _Float16 a = (_Float16)hrx;
    hrh[idx] = a; hrl[idx] = (_Float16)(hrx - (float)a);
    _Float16 c = (_Float16)hix;
    hih[idx] = c; hil[idx] = (_Float16)(hix - (float)c);
  }
}

// ---- ch = hs_re@C_re^T - hs_im@C_im^T + x@D^T;  y = LN1(x + ch) ---------
__global__ __launch_bounds__(256) void k_ch_ln1(
    const _Float16* __restrict__ hrh, const _Float16* __restrict__ hrl,
    const _Float16* __restrict__ hih, const _Float16* __restrict__ hil,
    const _Float16* __restrict__ xh, const _Float16* __restrict__ xl,
    const _Float16* __restrict__ Crh, const _Float16* __restrict__ Crl,
    const _Float16* __restrict__ Cnih, const _Float16* __restrict__ Cnil,
    const _Float16* __restrict__ Dh, const _Float16* __restrict__ Dl,
    const float* __restrict__ x, const float* __restrict__ ln1g,
    const float* __restrict__ ln1b,
    _Float16* __restrict__ yh, _Float16* __restrict__ yl) {
  // 6 staged planes (16 x HPAD halves each) + LN tile overlap + stats
  __shared__ __align__(16) unsigned char smem[6 * 16 * HPAD * 2 + 128];
  _Float16* s_hrh = (_Float16*)(smem);
  _Float16* s_hrl = s_hrh + 16 * HPAD;
  _Float16* s_hih = s_hrl + 16 * HPAD;
  _Float16* s_hil = s_hih + 16 * HPAD;
  _Float16* s_xh  = s_hil + 16 * HPAD;
  _Float16* s_xl  = s_xh + 16 * HPAD;
  float* sfin = (float*)(smem);            // overlaps staging (dead after GEMMs)
  float* smu = (float*)(smem + 6 * 16 * HPAD * 2);
  float* srs = smu + 16;

  const int tid = threadIdx.x;
  const int row0 = blockIdx.x * 16;
  {
    const size_t go = (size_t)row0 * 256;
    stage_plane(s_hrh, hrh + go, tid);
    stage_plane(s_hrl, hrl + go, tid);
    stage_plane(s_hih, hih + go, tid);
    stage_plane(s_hil, hil + go, tid);
    stage_plane(s_xh, xh + go, tid);
    stage_plane(s_xl, xl + go, tid);
  }
  __builtin_prefetch(Crh + (size_t)tid * 128, 0, 1);
  __builtin_prefetch(Cnih + (size_t)tid * 128, 0, 1);
  __builtin_prefetch(Dh + (size_t)tid * 128, 0, 1);
  cp_async_wait();
  __syncthreads();
  const int wave = tid >> 5, lane = tid & 31;
  const int hlf = lane >> 4, m = lane & 15;
  const int ncol0 = wave * 32;
  v8f acc[2] = {};
  gemm_acc<2, 8>(acc, s_hrh, s_hrl, HPAD, Crh, Crl, 256, ncol0, lane);
  gemm_acc<2, 8>(acc, s_hih, s_hil, HPAD, Cnih, Cnil, 256, ncol0, lane);
  gemm_acc<2, 8>(acc, s_xh, s_xl, HPAD, Dh, Dl, 256, ncol0, lane);
  __syncthreads();  // staged tiles dead; reuse region for LN tile
#pragma unroll
  for (int nt = 0; nt < 2; ++nt) {
    const int c = ncol0 + nt * 16 + m;
#pragma unroll
    for (int v = 0; v < 8; ++v) {
      const int r = v + 8 * hlf;
      sfin[r * 272 + c] = acc[nt][v] + x[(size_t)(row0 + r) * 256 + c];
    }
  }
  __syncthreads();
  if (tid < 16) {
    float s = 0.f, s2 = 0.f;
    for (int c = 0; c < 256; ++c) { float v = sfin[tid * 272 + c]; s += v; s2 += v * v; }
    const float mu = s * (1.0f / 256.0f);
    float var = s2 * (1.0f / 256.0f) - mu * mu;
    var = var < 0.f ? 0.f : var;
    smu[tid] = mu;
    srs[tid] = 1.0f / sqrtf(var + LN_EPS);
  }
  __syncthreads();
  for (int i = tid; i < 16 * 256; i += 256) {
    const int r = i >> 8, c = i & 255;
    const float v = (sfin[r * 272 + c] - smu[r]) * srs[r] * ln1g[c] + ln1b[c];
    _Float16 h = (_Float16)v;
    const size_t o = (size_t)(row0 + r) * 256 + c;
    yh[o] = h;
    yl[o] = (_Float16)(v - (float)h);
  }
}

// ---- fused FFN: z = gelu(y@w1^T+b1); out = LN2(y + z@w2^T + b2) ---------
__global__ __launch_bounds__(256) void k_ffn(
    const _Float16* __restrict__ yh, const _Float16* __restrict__ yl,
    const _Float16* __restrict__ w1h, const _Float16* __restrict__ w1l,
    const float* __restrict__ b1,
    const _Float16* __restrict__ w2h, const _Float16* __restrict__ w2l,
    const float* __restrict__ b2,
    const float* __restrict__ ln2g, const float* __restrict__ ln2b,
    float* __restrict__ out) {
  // y planes (2 x 16 x HPAD) | z planes (2 x 16 x ZPAD, overlapped later by LN tile)
  __shared__ __align__(16) unsigned char smem[2 * 16 * HPAD * 2 + 2 * 16 * ZPAD * 2 + 128];
  _Float16* syh_s = (_Float16*)smem;
  _Float16* syl_s = syh_s + 16 * HPAD;
  _Float16* szh_s = syl_s + 16 * HPAD;
  _Float16* szl_s = szh_s + 16 * ZPAD;
  float* sfin = (float*)szh_s;             // overlaps z region (dead after GEMM2)
  float* smu = (float*)(smem + 2 * 16 * HPAD * 2 + 2 * 16 * ZPAD * 2);
  float* srs = smu + 16;

  const int tid = threadIdx.x;
  const int row0 = blockIdx.x * 16;
  stage_plane(syh_s, yh + (size_t)row0 * 256, tid);
  stage_plane(syl_s, yl + (size_t)row0 * 256, tid);
  __builtin_prefetch(w1h + (size_t)tid * 1024, 0, 1);
  __builtin_prefetch(w2h + (size_t)tid * 1024, 0, 1);
  cp_async_wait();
  __syncthreads();
  const int wave = tid >> 5, lane = tid & 31;
  const int hlf = lane >> 4, m = lane & 15;
  {  // GEMM1: 16x1024, each wave 128 cols, K=256
    v8f acc[8] = {};
    const int nb = wave * 128;
    gemm_acc<8, 8>(acc, syh_s, syl_s, HPAD, w1h, w1l, 256, nb, lane);
#pragma unroll
    for (int nt = 0; nt < 8; ++nt) {
      const int c = nb + nt * 16 + m;
#pragma unroll
      for (int v = 0; v < 8; ++v) {
        const int r = v + 8 * hlf;
        float z = acc[nt][v] + b1[c];
        z = 0.5f * z * (1.0f + erff(z * 0.70710678118654752f));  // exact gelu
        _Float16 h = (_Float16)z;
        szh_s[r * ZPAD + c] = h;
        szl_s[r * ZPAD + c] = (_Float16)(z - (float)h);
      }
    }
  }
  __syncthreads();
  // GEMM2: 16x256, each wave 32 cols, K=1024
  v8f acc2[2] = {};
  const int ncol0 = wave * 32;
  gemm_acc<2, 32>(acc2, szh_s, szl_s, ZPAD, w2h, w2l, 1024, ncol0, lane);
  __syncthreads();  // z region dead; reuse for LN tile
#pragma unroll
  for (int nt = 0; nt < 2; ++nt) {
    const int c = ncol0 + nt * 16 + m;
#pragma unroll
    for (int v = 0; v < 8; ++v) {
      const int r = v + 8 * hlf;
      const float yv = (float)syh_s[r * HPAD + c] + (float)syl_s[r * HPAD + c];
      sfin[r * 272 + c] = acc2[nt][v] + b2[c] + yv;
    }
  }
  __syncthreads();
  if (tid < 16) {
    float s = 0.f, s2 = 0.f;
    for (int c = 0; c < 256; ++c) { float v = sfin[tid * 272 + c]; s += v; s2 += v * v; }
    const float mu = s * (1.0f / 256.0f);
    float var = s2 * (1.0f / 256.0f) - mu * mu;
    var = var < 0.f ? 0.f : var;
    smu[tid] = mu;
    srs[tid] = 1.0f / sqrtf(var + LN_EPS);
  }
  __syncthreads();
  for (int i = tid; i < 16 * 256; i += 256) {
    const int r = i >> 8, c = i & 255;
    out[(size_t)(row0 + r) * 256 + c] =
        (sfin[r * 272 + c] - smu[r]) * srs[r] * ln2g[c] + ln2b[c];
  }
}

// ---- launcher ------------------------------------------------------------
extern "C" void kernel_launch(void* const* d_in, const int* in_sizes, int n_in,
                              void* d_out, int out_size, void* d_ws, size_t ws_size,
                              hipStream_t stream) {
  const float* x = (const float*)d_in[0];
  const float* B_re = (const float*)d_in[1];
  const float* B_im = (const float*)d_in[2];
  const float* C_re = (const float*)d_in[3];
  const float* C_im = (const float*)d_in[4];
  const float* Dm = (const float*)d_in[5];
  const float* nu_log = (const float*)d_in[6];
  const float* theta_log = (const float*)d_in[7];
  const float* gamma_log = (const float*)d_in[8];
  const float* ln1g = (const float*)d_in[9];
  const float* ln1b = (const float*)d_in[10];
  const float* ln2g = (const float*)d_in[11];
  const float* ln2b = (const float*)d_in[12];
  const float* w1 = (const float*)d_in[13];
  const float* b1 = (const float*)d_in[14];
  const float* w2 = (const float*)d_in[15];
  const float* b2 = (const float*)d_in[16];
  float* outp = (float*)d_out;

  char* p = (char*)d_ws;
  auto alloc = [&](size_t bytes) {
    char* r = p;
    p += (bytes + 255) & ~(size_t)255;
    return r;
  };
  _Float16* xh = (_Float16*)alloc(NX * 2);
  _Float16* xl = (_Float16*)alloc(NX * 2);
  float* bx_re = (float*)alloc(NX * 4);
  float* bx_im = (float*)alloc(NX * 4);
  _Float16* hrh = (_Float16*)alloc(NX * 2);
  _Float16* hrl = (_Float16*)alloc(NX * 2);
  _Float16* hih = (_Float16*)alloc(NX * 2);
  _Float16* hil = (_Float16*)alloc(NX * 2);
  _Float16* yh = (_Float16*)alloc(NX * 2);
  _Float16* yl = (_Float16*)alloc(NX * 2);
  const size_t HH = (size_t)H_DIM * H_DIM;        // 65536
  const size_t HF = (size_t)H_DIM * FFN_DIM;      // 262144
  _Float16* Breh = (_Float16*)alloc(HH * 2);
  _Float16* Brel = (_Float16*)alloc(HH * 2);
  _Float16* Bimh = (_Float16*)alloc(HH * 2);
  _Float16* Biml = (_Float16*)alloc(HH * 2);
  _Float16* Crh = (_Float16*)alloc(HH * 2);
  _Float16* Crl = (_Float16*)alloc(HH * 2);
  _Float16* Cnih = (_Float16*)alloc(HH * 2);
  _Float16* Cnil = (_Float16*)alloc(HH * 2);
  _Float16* Dh = (_Float16*)alloc(HH * 2);
  _Float16* Dl = (_Float16*)alloc(HH * 2);
  _Float16* w1h = (_Float16*)alloc(HF * 2);
  _Float16* w1l = (_Float16*)alloc(HF * 2);
  _Float16* w2h = (_Float16*)alloc(HF * 2);
  _Float16* w2l = (_Float16*)alloc(HF * 2);

  // precision splits
  k_split<<<(unsigned)((NX + 255) / 256), 256, 0, stream>>>(x, xh, xl, (int)NX, 1.0f);
  k_split<<<(unsigned)((HH + 255) / 256), 256, 0, stream>>>(B_re, Breh, Brel, (int)HH, 1.0f);
  k_split<<<(unsigned)((HH + 255) / 256), 256, 0, stream>>>(B_im, Bimh, Biml, (int)HH, 1.0f);
  k_split<<<(unsigned)((HH + 255) / 256), 256, 0, stream>>>(C_re, Crh, Crl, (int)HH, 1.0f);
  k_split<<<(unsigned)((HH + 255) / 256), 256, 0, stream>>>(C_im, Cnih, Cnil, (int)HH, -1.0f);
  k_split<<<(unsigned)((HH + 255) / 256), 256, 0, stream>>>(Dm, Dh, Dl, (int)HH, 1.0f);
  k_split<<<(unsigned)((HF + 255) / 256), 256, 0, stream>>>(w1, w1h, w1l, (int)HF, 1.0f);
  k_split<<<(unsigned)((HF + 255) / 256), 256, 0, stream>>>(w2, w2h, w2l, (int)HF, 1.0f);

  // input projections (WMMA)
  k_gemm_bx<<<NROWS / 16, 256, 0, stream>>>(xh, xl, Breh, Brel, Bimh, Biml, bx_re, bx_im);
  // sequential recurrence
  k_scan<<<(B_DIM * H_DIM) / 256, 256, 0, stream>>>(bx_re, bx_im, nu_log, theta_log,
                                                    gamma_log, hrh, hrl, hih, hil);
  // output projections + residual + LN1 (WMMA)
  k_ch_ln1<<<NROWS / 16, 256, 0, stream>>>(hrh, hrl, hih, hil, xh, xl, Crh, Crl,
                                           Cnih, Cnil, Dh, Dl, x, ln1g, ln1b, yh, yl);
  // fused FFN + residual + LN2 (WMMA)
  k_ffn<<<NROWS / 16, 256, 0, stream>>>(yh, yl, w1h, w1l, b1, w2h, w2l, b2,
                                        ln2g, ln2b, outp);
}
